// AdditiveAttention_14568529068259
// MI455X (gfx1250) — matrix operand order repaired and checked by reading, not compiled, not run on hardware
//
#include <hip/hip_runtime.h>
#include <stdint.h>

#define VDIM  512
#define D2    1024
#define BATCH 64
#define SEQ   1024
#define ROWS  (BATCH * SEQ)   // 65536
#define TROWS 64              // rows per workgroup in the WMMA kernel

typedef __attribute__((ext_vector_type(16))) __bf16        bf16x16;
typedef __attribute__((ext_vector_type(8)))  float         floatx8;
typedef __attribute__((ext_vector_type(4)))  unsigned int  uintx4;

union FragBF {
    bf16x16 v;
    uintx4  u2[2];
};

__device__ __forceinline__ uint16_t f32_to_bf16_rne(float x) {
    uint32_t u = __float_as_uint(x);
    u += 0x7FFFu + ((u >> 16) & 1u);
    return (uint16_t)(u >> 16);
}

#define WMMA_BF16(A, B, C) \
    __builtin_amdgcn_wmma_f32_16x16x32_bf16(false, (A).v, false, (B).v, (short)0, (C), false, false)

// -------------------------------------------------------------------------
// Kernel 1: W_h f32 -> bf16 (row-major [1024][1024])
// -------------------------------------------------------------------------
__global__ void k_convert_wh(const float* __restrict__ Wh, uint16_t* __restrict__ whbf) {
    int idx = (blockIdx.x * blockDim.x + threadIdx.x) * 4;
    float4 f = *reinterpret_cast<const float4*>(Wh + idx);
    whbf[idx + 0] = f32_to_bf16_rne(f.x);
    whbf[idx + 1] = f32_to_bf16_rne(f.y);
    whbf[idx + 2] = f32_to_bf16_rne(f.z);
    whbf[idx + 3] = f32_to_bf16_rne(f.w);
}

// -------------------------------------------------------------------------
// Kernel 2: state_feature[b,e] = b_s[e] + [h(b),c(b)] . W_s[e,:]
// -------------------------------------------------------------------------
__global__ void k_state_feature(const float* __restrict__ h, const float* __restrict__ c,
                                const float* __restrict__ Ws, const float* __restrict__ bs,
                                float* __restrict__ sf) {
    int gid = blockIdx.x * blockDim.x + threadIdx.x;   // 0..65535
    int b = gid & (BATCH - 1);
    int e = gid >> 6;
    const float* wr = Ws + (size_t)e * D2;
    float acc = bs[e];
    for (int d = 0; d < VDIM; ++d) acc += h[b * VDIM + d] * wr[d];
    for (int d = 0; d < VDIM; ++d) acc += c[b * VDIM + d] * wr[VDIM + d];
    sf[b * D2 + e] = acc;
}

// -------------------------------------------------------------------------
// Kernel 3: fused encoder_feature GEMM (bf16 WMMA) + tanh + score reduction
// grid = 1024 workgroups of 128 threads (4 waves); each WG owns 64 rows.
// LDS: 64x1024 bf16 A-tile (128KB) + score[64] + cov[64].
// Each wave handles 256 output columns as 8 passes of TWO N-tiles, so the
// four A fragments loaded per k-step feed EIGHT WMMAs (2x LDS reuse).
// -------------------------------------------------------------------------
__global__ __launch_bounds__(128, 1)
void k_score(const float* __restrict__ ctx, const float* __restrict__ coverage,
             const uint16_t* __restrict__ whbf, const float* __restrict__ sf,
             const float* __restrict__ Wx, const float* __restrict__ vvec,
             float* __restrict__ score_out) {
    extern __shared__ __align__(16) char smem_raw[];
    uint16_t* a_lds     = reinterpret_cast<uint16_t*>(smem_raw);                 // 64*1024 bf16
    float*    score_lds = reinterpret_cast<float*>(smem_raw + TROWS * D2 * 2);   // 64
    float*    cov_lds   = score_lds + TROWS;                                     // 64

    const int tid  = threadIdx.x;
    const int row0 = blockIdx.x * TROWS;        // tile never crosses a batch (1024 % 64 == 0)
    const int bidx = row0 >> 10;                // row0 / SEQ

    // ---- stage 64 context rows f32 -> bf16 into LDS ----
    const float4* src = reinterpret_cast<const float4*>(ctx + (size_t)row0 * D2);
    #pragma unroll 4
    for (int it = 0; it < (TROWS * D2 / 4) / 128; ++it) {
        int i4 = it * 128 + tid;
        float4 f = src[i4];
        int e = i4 * 4;
        a_lds[e + 0] = f32_to_bf16_rne(f.x);
        a_lds[e + 1] = f32_to_bf16_rne(f.y);
        a_lds[e + 2] = f32_to_bf16_rne(f.z);
        a_lds[e + 3] = f32_to_bf16_rne(f.w);
    }
    if (tid < TROWS) {
        score_lds[tid] = 0.0f;
        cov_lds[tid]   = coverage[row0 + tid];
    }
    __syncthreads();

    const int lane = tid & 31;
    const int wave = tid >> 5;
    const int l15  = lane & 15;
    const int up   = (lane >> 4) & 1;
    const int kbA  = up ? 8  : 0;   // 16-bit A frag: upper half-wave holds K+8 group
    const int kbB  = up ? 8  : 0;   // 16-bit B frag: upper half-wave holds K+8 group

    const float* sfr = sf + (size_t)bidx * D2;

    float sacc[4][8];
    #pragma unroll
    for (int r = 0; r < 4; ++r)
        #pragma unroll
        for (int i = 0; i < 8; ++i) sacc[r][i] = 0.0f;

    const floatx8 vzero = {0.f, 0.f, 0.f, 0.f, 0.f, 0.f, 0.f, 0.f};

    // each wave owns 256 consecutive output columns; 8 passes of 2 N-tiles
    for (int t = 0; t < 8; ++t) {
        const int n0    = wave * 256 + t * 32;
        const int ncol0 = n0 + l15;
        const int ncol1 = n0 + 16 + l15;
        const float sfv0 = sfr[ncol0],  sfv1 = sfr[ncol1];
        const float wxv0 = Wx[ncol0],   wxv1 = Wx[ncol1];
        const float vv0  = vvec[ncol0], vv1  = vvec[ncol1];

        floatx8 c00 = vzero, c01 = vzero, c02 = vzero, c03 = vzero;   // ncol0, row tiles 0..3
        floatx8 c10 = vzero, c11 = vzero, c12 = vzero, c13 = vzero;   // ncol1, row tiles 0..3

        // LDS base pointers: row tiles 0/1 off ap01 (DS offsets 0 / 32768),
        // row tiles 2/3 off ap23 (same immediate offsets) -> no per-iter v_adds
        const uint16_t* ap01 = a_lds + l15 * D2 + kbA;
        const uint16_t* ap23 = ap01 + 32 * D2;
        const uint16_t* bp0  = whbf + (size_t)ncol0 * D2 + kbB;   // B[k][n] = W_h[n][k]
        const uint16_t* bp1  = bp0 + 16 * D2;

        for (int ks = 0; ks < 32; ++ks) {
            __builtin_prefetch(bp0 + 32, 0, 3);
            __builtin_prefetch(bp1 + 32, 0, 3);

            FragBF b0, b1;   // B 32x16 bf16: lane = column, K groups of 8 (same order as A)
            b0.u2[0] = *reinterpret_cast<const uintx4*>(bp0);
            b0.u2[1] = *reinterpret_cast<const uintx4*>(bp0 + 16);
            b1.u2[0] = *reinterpret_cast<const uintx4*>(bp1);
            b1.u2[1] = *reinterpret_cast<const uintx4*>(bp1 + 16);

            FragBF a0, a1, a2, a3;   // four A row-tiles, loaded once, used twice
            a0.u2[0] = *reinterpret_cast<const uintx4*>(ap01);
            a0.u2[1] = *reinterpret_cast<const uintx4*>(ap01 + 16);
            a1.u2[0] = *reinterpret_cast<const uintx4*>(ap01 + 16 * D2);
            a1.u2[1] = *reinterpret_cast<const uintx4*>(ap01 + 16 * D2 + 16);
            a2.u2[0] = *reinterpret_cast<const uintx4*>(ap23);
            a2.u2[1] = *reinterpret_cast<const uintx4*>(ap23 + 16);
            a3.u2[0] = *reinterpret_cast<const uintx4*>(ap23 + 16 * D2);
            a3.u2[1] = *reinterpret_cast<const uintx4*>(ap23 + 16 * D2 + 16);

            c00 = WMMA_BF16(a0, b0, c00);
            c10 = WMMA_BF16(a0, b1, c10);
            c01 = WMMA_BF16(a1, b0, c01);
            c11 = WMMA_BF16(a1, b1, c11);
            c02 = WMMA_BF16(a2, b0, c02);
            c12 = WMMA_BF16(a2, b1, c12);
            c03 = WMMA_BF16(a3, b0, c03);
            c13 = WMMA_BF16(a3, b1, c13);

            ap01 += 32;
            ap23 += 32;
            bp0  += 32;
            bp1  += 32;
        }

        // fused epilogue: e = tanh(gemm + sf + cov*Wx); score += e * v
        #pragma unroll
        for (int i = 0; i < 8; ++i) {
            int m = up * 8 + i;   // D layout: VGPR i -> row i (+8 for upper half-wave)
            sacc[0][i] += tanhf(c00[i] + sfv0 + cov_lds[m +  0] * wxv0) * vv0
                        + tanhf(c10[i] + sfv1 + cov_lds[m +  0] * wxv1) * vv1;
            sacc[1][i] += tanhf(c01[i] + sfv0 + cov_lds[m + 16] * wxv0) * vv0
                        + tanhf(c11[i] + sfv1 + cov_lds[m + 16] * wxv1) * vv1;
            sacc[2][i] += tanhf(c02[i] + sfv0 + cov_lds[m + 32] * wxv0) * vv0
                        + tanhf(c12[i] + sfv1 + cov_lds[m + 32] * wxv1) * vv1;
            sacc[3][i] += tanhf(c03[i] + sfv0 + cov_lds[m + 48] * wxv0) * vv0
                        + tanhf(c13[i] + sfv1 + cov_lds[m + 48] * wxv1) * vv1;
        }
    }

    // reduce per-row partial scores across lanes/waves via LDS float atomics
    #pragma unroll
    for (int r = 0; r < 4; ++r)
        #pragma unroll
        for (int i = 0; i < 8; ++i)
            atomicAdd(&score_lds[r * 16 + up * 8 + i], sacc[r][i]);
    __syncthreads();

    if (tid < TROWS) score_out[row0 + tid] = score_lds[tid];
}

// -------------------------------------------------------------------------
// Kernel 4: masked softmax + renormalization per batch row, plus coverage copy
// -------------------------------------------------------------------------
__global__ void k_softmax(const float* __restrict__ score, const float* __restrict__ mask,
                          const float* __restrict__ coverage,
                          float* __restrict__ att_out, float* __restrict__ cov_out) {
    __shared__ float red[256];
    const int b = blockIdx.x, tid = threadIdx.x;
    const float* s = score + b * SEQ;

    float sv[4], mv[4];
    float mx = -1e30f;
    #pragma unroll
    for (int j = 0; j < 4; ++j) {
        int l = tid + j * 256;
        sv[j] = s[l];
        mv[j] = mask[b * SEQ + l];
        mx = fmaxf(mx, sv[j]);
    }
    red[tid] = mx; __syncthreads();
    for (int off = 128; off > 0; off >>= 1) {
        if (tid < off) red[tid] = fmaxf(red[tid], red[tid + off]);
        __syncthreads();
    }
    mx = red[0]; __syncthreads();

    float ev[4], z = 0.0f;
    #pragma unroll
    for (int j = 0; j < 4; ++j) { ev[j] = __expf(sv[j] - mx) * mv[j]; z += ev[j]; }
    red[tid] = z; __syncthreads();
    for (int off = 128; off > 0; off >>= 1) {
        if (tid < off) red[tid] += red[tid + off];
        __syncthreads();
    }
    const float inv = 1.0f / red[0];

    #pragma unroll
    for (int j = 0; j < 4; ++j) {
        int l = tid + j * 256;
        att_out[b * SEQ + l] = ev[j] * inv;
        cov_out[b * SEQ + l] = coverage[b * SEQ + l];   // pass-through output
    }
}

// -------------------------------------------------------------------------
// Kernel 5: weighted_context[b,e] = sum_l attn[b,l] * context[b,l,e]
// -------------------------------------------------------------------------
__global__ void k_weighted_ctx(const float* __restrict__ att, const float* __restrict__ ctx,
                               float* __restrict__ wc) {
    const int b = blockIdx.y;
    const int e = blockIdx.x * 256 + threadIdx.x;
    const float* a = att + b * SEQ;
    const float* c = ctx + (size_t)b * SEQ * D2 + e;
    float acc = 0.0f;
    #pragma unroll 4
    for (int l = 0; l < SEQ; ++l) acc += a[l] * c[(size_t)l * D2];
    wc[b * D2 + e] = acc;
}

// -------------------------------------------------------------------------
extern "C" void kernel_launch(void* const* d_in, const int* in_sizes, int n_in,
                              void* d_out, int out_size, void* d_ws, size_t ws_size,
                              hipStream_t stream) {
    const float* h    = (const float*)d_in[0];   // [1,64,512]
    const float* c    = (const float*)d_in[1];   // [1,64,512]
    const float* ctx  = (const float*)d_in[2];   // [64,1024,1024]
    const float* cov  = (const float*)d_in[3];   // [64,1024]
    const float* mask = (const float*)d_in[4];   // [64,1024]
    const float* Wh   = (const float*)d_in[5];   // [1024,1024]
    const float* Ws   = (const float*)d_in[6];   // [1024,1024]
    const float* bs   = (const float*)d_in[7];   // [1024]
    const float* Wx   = (const float*)d_in[8];   // [1024,1]
    const float* v    = (const float*)d_in[9];   // [1,1024]

    float* out       = (float*)d_out;
    float* wc_out    = out;                      // weighted_context  [64,1,1024]
    float* att_out   = out + 65536;              // normed attention  [64,1024]
    float* cov_out   = out + 131072;             // coverage          [64,1024]
    float* score_out = out + 196608;             // score             [64,1024]

    uint16_t* whbf = (uint16_t*)d_ws;                             // 2 MB
    float*    sf   = (float*)((char*)d_ws + (size_t)D2 * D2 * 2); // 256 KB

    k_convert_wh<<<(D2 * D2) / 1024, 256, 0, stream>>>(Wh, whbf);
    k_state_feature<<<(BATCH * D2) / 256, 256, 0, stream>>>(h, c, Ws, bs, sf);

    const size_t smem = (size_t)TROWS * D2 * 2 + (size_t)TROWS * 4 * 2;  // 128KB + 512B
    k_score<<<ROWS / TROWS, 128, smem, stream>>>(ctx, cov, whbf, sf, Wx, v, score_out);

    k_softmax<<<BATCH, 256, 0, stream>>>(score_out, mask, cov, att_out, cov_out);
    k_weighted_ctx<<<dim3(4, BATCH), 256, 0, stream>>>(att_out, ctx, wc_out);
}
